// TensorAttention_14422500180267
// MI455X (gfx1250) — matrix operand order repaired and checked by reading, not compiled, not run on hardware
//
#include <hip/hip_runtime.h>
#include <hip/hip_bf16.h>

// Problem constants (match reference)
#define B_   8
#define TD1  128
#define TD2  128   // words per clause (softmax axis)
#define WD   512   // word dim (K of GEMM)
#define PROJ 256   // proj dim (N of GEMM)

// Flip to 0 if the async-to-LDS inline asm fails to assemble.
#define USE_ASYNC_COPY 1

typedef __attribute__((ext_vector_type(16))) __bf16 v16bf;
typedef __attribute__((ext_vector_type(8)))  float  v8f;

// LDS row pitch: 512 + 8 pad bf16 elems -> 1040B rows (260 dwords, %64 = 4
// banks/row) so 16-lane b128 fragment reads rotate across all 64 banks.
#define XPITCH 520

union FragU { uint4 u[2]; v16bf v; };

// Kernel 0: att_proj (WD x PROJ fp32, row-major) -> Wt (PROJ x WD bf16).
// Transposed so each lane's B-matrix WMMA fragment is two 16B runs of one Wt row.
__global__ __launch_bounds__(256)
void ta_prep_wt(const float* __restrict__ W, __bf16* __restrict__ Wt) {
  const int d = blockIdx.x;    // 0..WD-1
  const int p = threadIdx.x;   // 0..PROJ-1 (coalesced read over p)
  Wt[p * WD + d] = (__bf16)W[d * PROJ + p];
}

// Fused kernel: one workgroup per (b, c) clause tile; 8 waves, wave = M-tile.
__global__ __launch_bounds__(256)
void ta_fused(const float* __restrict__ X, const __bf16* __restrict__ Wt,
              const float* __restrict__ scorer, float* __restrict__ out) {
  // One 130KB buffer, used twice:
  //   Phase A: bf16 X tile (128 rows x 512 K) -> A fragments hoisted to VGPRs
  //   Phase B: two staged 128-row chunks of Wt (B operand, shared by 8 waves)
  __shared__ __align__(16) __bf16 SMEM[TD2 * XPITCH];
  __shared__ float scorer_l[PROJ];
  __shared__ float att_l[TD2];
  __shared__ float red[256];

  const int t    = threadIdx.x;
  const int lane = t & 31;
  const int wave = t >> 5;        // 8 waves; wave == M-tile index
  const int half = lane >> 4;     // lane half selects K sub-range (ISA layout)
  const int l15  = lane & 15;

  const long tileBase = (long)blockIdx.x * (TD2 * WD);
  const float* Xt = X + tileBase;

  // ---- Phase A: stream X tile (fp32, coalesced float4) -> LDS as bf16 ----
  #pragma unroll
  for (int i = 0; i < (TD2 * WD) / (256 * 4); ++i) {   // 64 float4 / thread
    const int idx = (t + i * 256) * 4;
    const int row = idx >> 9;          // /512
    const int col = idx & (WD - 1);
    const float4 v = *(const float4*)(Xt + idx);
    __bf16* dst = &SMEM[row * XPITCH + col];
    dst[0] = (__bf16)v.x; dst[1] = (__bf16)v.y;
    dst[2] = (__bf16)v.z; dst[3] = (__bf16)v.w;
  }
  scorer_l[t] = scorer[t];             // PROJ == blockDim
  __syncthreads();

  // A 16x32 bf16 fragments for this wave's M-tile, kept in VGPRs (16 x 32B):
  // lane<16 -> K {k0..k0+7, k0+16..k0+23}; lane>=16 -> +8 offsets.
  const int mrow = wave * 16 + l15;
  FragU areg[WD / 32];
  #pragma unroll
  for (int kk = 0; kk < WD / 32; ++kk) {
    const int ca = kk * 32 + half * 8;
    areg[kk].u[0] = *(const uint4*)(&SMEM[mrow * XPITCH + ca]);
    areg[kk].u[1] = *(const uint4*)(&SMEM[mrow * XPITCH + ca + 16]);
  }

  // ---- Phase B: scores[m] = sum_p tanh((X·Wp)[m,p]) * scorer[p] ----
  float sacc[8] = {0.f, 0.f, 0.f, 0.f, 0.f, 0.f, 0.f, 0.f};

  for (int chunk = 0; chunk < 2; ++chunk) {
    __syncthreads();   // all waves done reading SMEM's previous contents

    // Cooperative, fully-coalesced copy of 128 Wt rows (128KB) into SMEM
    // (padded pitch). One b128 per iteration per thread.
    const uint4* src = (const uint4*)(Wt + chunk * 128 * WD);
    #pragma unroll
    for (int i = 0; i < (128 * WD * 2) / (256 * 16); ++i) {   // 32 iters
      const int u   = t + i * 256;        // 16B-granule index
      const int row = u >> 6;             // 64 granules per 1KB source row
      const int c16 = u & 63;
#if USE_ASYNC_COPY
      __attribute__((address_space(3))) __bf16* lp =
          (__attribute__((address_space(3))) __bf16*)&SMEM[row * XPITCH + c16 * 8];
      asm volatile("global_load_async_to_lds_b128 %0, %1, off"
                   :: "v"(lp), "v"(src + u) : "memory");
#else
      const uint4 v = src[u];
      *(uint4*)(&SMEM[row * XPITCH + c16 * 8]) = v;
#endif
    }
#if USE_ASYNC_COPY
    asm volatile("s_wait_asynccnt 0x0" ::: "memory");
#endif
    __syncthreads();

    // Each wave: its M-tile x the 8 N-tiles of this chunk, B from LDS.
    for (int j = 0; j < 8; ++j) {
      const int nloc = j * 16 + l15;      // local Wt row == N column
      v8f c = {0.f, 0.f, 0.f, 0.f, 0.f, 0.f, 0.f, 0.f};
      #pragma unroll
      for (int kk = 0; kk < WD / 32; ++kk) {
        // B 32x16 bf16: lane<16 -> K {0..7, 16..23}; lane>=16 -> +8 offsets.
        const __bf16* bp = &SMEM[nloc * XPITCH + kk * 32 + half * 8];
        FragU b;
        b.u[0] = ((const uint4*)bp)[0];
        b.u[1] = ((const uint4*)bp)[2];
        c = __builtin_amdgcn_wmma_f32_16x16x32_bf16(
              false, areg[kk].v, false, b.v, (short)0, c, false, false);
      }
      // C layout: VGPR r, lane -> (M = r + 8*half, N = chunk*128 + j*16 + l15)
      const float sc = scorer_l[chunk * 128 + j * 16 + l15];
      #pragma unroll
      for (int r = 0; r < 8; ++r) sacc[r] += tanhf(c[r]) * sc;
    }
  }

  // Reduce over N across each 16-lane half (butterfly stays within the half)
  #pragma unroll
  for (int r = 0; r < 8; ++r) {
    float v = sacc[r];
    v += __shfl_xor(v, 1, 32);
    v += __shfl_xor(v, 2, 32);
    v += __shfl_xor(v, 4, 32);
    v += __shfl_xor(v, 8, 32);
    sacc[r] = v;
  }
  if (l15 == 0) {
    #pragma unroll
    for (int r = 0; r < 8; ++r)
      att_l[wave * 16 + half * 8 + r] = sacc[r];   // raw logits, 128 rows
  }
  __syncthreads();

  // ---- Phase C: softmax over the 128 logits ----
  const float v0 = (t < TD2) ? att_l[t] : -3.4e38f;
  red[t] = v0;
  __syncthreads();
  for (int s = 128; s > 0; s >>= 1) {
    if (t < s) red[t] = fmaxf(red[t], red[t + s]);
    __syncthreads();
  }
  const float mx = red[0];
  __syncthreads();
  const float e = (t < TD2) ? __expf(v0 - mx) : 0.f;
  red[t] = e;
  __syncthreads();
  for (int s = 128; s > 0; s >>= 1) {
    if (t < s) red[t] += red[t + s];
    __syncthreads();
  }
  const float inv = 1.f / red[0];
  __syncthreads();
  if (t < TD2) att_l[t] = e * inv;
  __syncthreads();

  // ---- Phase D: out[d] = sum_w att[w] * X[w,d] (exact fp32, L2-hot) ----
  float acc0 = 0.f, acc1 = 0.f;
  const int d0 = 2 * t;
  for (int w = 0; w < TD2; ++w) {
    const float aw = att_l[w];
    const float2 xv = *(const float2*)(Xt + w * WD + d0);
    acc0 += aw * xv.x;
    acc1 += aw * xv.y;
  }
  float* o = out + (long)blockIdx.x * WD;
  *(float2*)(o + d0) = make_float2(acc0, acc1);
}

extern "C" void kernel_launch(void* const* d_in, const int* in_sizes, int n_in,
                              void* d_out, int out_size, void* d_ws, size_t ws_size,
                              hipStream_t stream) {
  const float* X      = (const float*)d_in[0];  // (8,128,128,512) fp32
  const float* W      = (const float*)d_in[1];  // (512,256) fp32
  const float* scorer = (const float*)d_in[2];  // (256,) fp32
  float* out = (float*)d_out;                   // (8,128,512) fp32
  __bf16* Wt = (__bf16*)d_ws;                   // 256 KB bf16 transposed proj

  ta_prep_wt<<<WD, 256, 0, stream>>>(W, Wt);
  ta_fused<<<B_ * TD1, 256, 0, stream>>>(X, Wt, scorer, out);
}
